// MultiHeadAttnGRU_78451872629031
// MI455X (gfx1250) — matrix operand order repaired and checked by reading, not compiled, not run on hardware
//
#include <hip/hip_runtime.h>
#include <hip/hip_bf16.h>
#include <math.h>

// ---------------------------------------------------------------------------
// Problem constants (from reference)
// ---------------------------------------------------------------------------
#define LM   128      // memory length
#define BS   16       // batch
#define DD   512      // memory feature dim
#define HH   512      // hidden
#define VV   16000    // vocab
#define TT   32       // max_output_len
#define NHEAD 8
#define DK   64       // head dim

typedef _Float16 h16;
typedef __attribute__((ext_vector_type(16))) _Float16 v16h;
typedef __attribute__((ext_vector_type(8)))  _Float16 v8h;
typedef __attribute__((ext_vector_type(8)))  float    v8f;

__device__ __forceinline__ h16 f2h(float x) { return (h16)x; }

// ---------------------------------------------------------------------------
// WMMA fragment loaders (CDNA5 16x16x32 f16, wave32; contiguous k order 16*half + i, same in A and B)
// ---------------------------------------------------------------------------
// A (16xK tile, row-major source, row m contiguous in K):
//   lane l: m = l%16, half = l/16
//   element i -> K = k0 + 16*half + i
__device__ __forceinline__ v16h load_a(const h16* A, int lda, int m0, int k0) {
    const int lane = threadIdx.x & 31;
    const int half = lane >> 4;
    const int m    = (lane & 15) + m0;
    const h16* base = A + (size_t)m * lda + k0;
    return *(const v16h*)(base + 16 * half);
}

// B (Kx16 tile), source pre-packed COLUMN-major: Bt[n][k] contiguous in k.
//   lane l: n = l%16, half = l/16, element i -> K = k0 + 16*half + i
//   -> one contiguous 32-byte load per lane.
__device__ __forceinline__ v16h load_b(const h16* Bt, int ldb, int n0, int k0) {
    const int lane = threadIdx.x & 31;
    const int half = lane >> 4;
    const int n    = (lane & 15) + n0;
    return *(const v16h*)(Bt + (size_t)n * ldb + k0 + 16 * half);
}

// K-loop over NT adjacent 16x16 output tiles sharing one A fragment per step,
// hand-unrolled by 2 with sched_group_barrier pipeline hints:
//   [a0+b0 loads][2 WMMA][a1+half b1 loads][2 WMMA][rest b1 loads][4 WMMA]
// This forces machine-order loads-ahead-of-WMMAs (distinct fragment regs ->
// partial s_wait_loadcnt instead of 0x0 before every WMMA).
// Requires K % 64 == 0 (true for all call sites: 512 / 1024).
template <int NT>
__device__ __forceinline__ void wmma_tile_nt(const h16* A, int lda, int m0,
                                             const h16* Bt, int ldb, int n0,
                                             int K, v8f* c) {
    for (int k = 0; k < K; k += 64) {
        v16h a0 = load_a(A, lda, m0, k);
        v16h b0[NT];
#pragma unroll
        for (int j = 0; j < NT; ++j) b0[j] = load_b(Bt, ldb, n0 + 16 * j, k);
        v16h a1 = load_a(A, lda, m0, k + 32);
        v16h b1[NT];
#pragma unroll
        for (int j = 0; j < NT; ++j) b1[j] = load_b(Bt, ldb, n0 + 16 * j, k + 32);
#pragma unroll
        for (int j = 0; j < NT; ++j)
            c[j] = __builtin_amdgcn_wmma_f32_16x16x32_f16(
                       false, a0, false, b0[j], (short)0, c[j], false, false);
#pragma unroll
        for (int j = 0; j < NT; ++j)
            c[j] = __builtin_amdgcn_wmma_f32_16x16x32_f16(
                       false, a1, false, b1[j], (short)0, c[j], false, false);
        // pipeline shape: 0x0020 = VMEM read group, 0x0008 = MFMA/WMMA group
        __builtin_amdgcn_sched_group_barrier(0x0020, 2 + 2 * NT, 0); // a0 + b0[*]
        __builtin_amdgcn_sched_group_barrier(0x0008, 2, 0);
        __builtin_amdgcn_sched_group_barrier(0x0020, 1 + NT, 0);     // a1 + half b1
        __builtin_amdgcn_sched_group_barrier(0x0008, 2, 0);
        __builtin_amdgcn_sched_group_barrier(0x0020, 1 + NT, 0);     // rest of b1
        __builtin_amdgcn_sched_group_barrier(0x0008, NT, 0);
    }
}

// ---------------------------------------------------------------------------
// Prep kernels (once per launch)
// ---------------------------------------------------------------------------
// Wt[n*K + k] = (h16) W[k*N + n]   (fp32 row-major -> f16 column-major)
__global__ void transpose_f16_kernel(const float* __restrict__ W,
                                     h16* __restrict__ Wt, int K, int N) {
    size_t i = (size_t)blockIdx.x * blockDim.x + threadIdx.x;
    if (i >= (size_t)K * N) return;
    int    k = (int)(i % K);
    size_t n = i / K;
    Wt[n * (size_t)K + k] = f2h(W[(size_t)k * N + n]);
}

__global__ void convert_f16_kernel(const float* __restrict__ X,
                                   h16* __restrict__ Y, size_t n) {
    size_t i = (size_t)blockIdx.x * blockDim.x + threadIdx.x;
    if (i < n) Y[i] = f2h(X[i]);
}

__global__ void zero_state_kernel(float* __restrict__ logits, float* __restrict__ hid) {
    int i = blockIdx.x * 256 + threadIdx.x;
    if (i < BS * VV) logits[i] = 0.f;
    if (i < BS * HH) hid[i]    = 0.f;
}

// Hoisted K/V projection: Kp/Vp[(l*BS+b)*HH + n] = memory @ Wk/Wv + bias
// M = LM*BS = 2048, K = DD = 512, N = HH = 512.
// 2 mats x 128 M-tiles x 8 N-groups (4 tiles each) = 2048 waves.
__global__ void kv_proj_kernel(const h16* __restrict__ mem_h,
                               const h16* __restrict__ Wk_t, const h16* __restrict__ Wv_t,
                               const float* __restrict__ bk, const float* __restrict__ bv,
                               float* __restrict__ Kp, float* __restrict__ Vp) {
    int w   = blockIdx.x * 4 + (threadIdx.x >> 5);
    int mat = w >> 10;              // 0 -> K, 1 -> V (1024 waves each)
    int rem = w & 1023;
    int mt  = rem >> 3;             // 128 M tiles
    int ng  = rem & 7;              // 8 N groups of 64 columns
    const h16*   Bt   = mat ? Wv_t : Wk_t;
    const float* bias = mat ? bv   : bk;
    float*       Out  = mat ? Vp   : Kp;
    v8f c[4] = {};
    wmma_tile_nt<4>(mem_h, DD, mt * 16, Bt, DD, ng * 64, DD, c);
    const int lane = threadIdx.x & 31, half = lane >> 4;
#pragma unroll
    for (int j = 0; j < 4; ++j) {
        int n = ng * 64 + 16 * j + (lane & 15);
#pragma unroll
        for (int r = 0; r < 8; ++r) {
            int m = mt * 16 + r + 8 * half;
            Out[(size_t)m * HH + n] = c[j][r] + bias[n];
        }
    }
}

// ---------------------------------------------------------------------------
// Per-step kernels
// ---------------------------------------------------------------------------
// S1: argmax(prev logits) -> embedding gather -> query = [emb | hid] in f16
__global__ void argmax_embed_kernel(const float* __restrict__ logits,
                                    const float* __restrict__ emb,
                                    const float* __restrict__ hid,
                                    h16* __restrict__ query_h) {
    const int b = blockIdx.x, tid = threadIdx.x;       // 256 threads
    __shared__ float sv[256];
    __shared__ int   si[256];
    float best = -3.402823466e+38f; int bidx = 0;
    for (int v = tid; v < VV; v += 256) {
        float x = logits[b * VV + v];
        if (x > best) { best = x; bidx = v; }
    }
    sv[tid] = best; si[tid] = bidx;
    __syncthreads();
    for (int off = 128; off; off >>= 1) {
        if (tid < off) {
            float ov = sv[tid + off]; int oi = si[tid + off];
            if (ov > sv[tid] || (ov == sv[tid] && oi < si[tid])) { sv[tid] = ov; si[tid] = oi; }
        }
        __syncthreads();
    }
    const int idx = si[0];
    for (int j = tid; j < HH; j += 256) {
        query_h[b * (2 * HH) + j]      = f2h(emb[(size_t)idx * HH + j]);
        query_h[b * (2 * HH) + HH + j] = f2h(hid[b * HH + j]);
    }
}

// S2: qh = query @ Wq + bq   (M=16, K=1024, N=512) -> 8 waves x 4 tiles
__global__ void qproj_kernel(const h16* __restrict__ query_h,
                             const h16* __restrict__ Wq_t,
                             const float* __restrict__ bq,
                             float* __restrict__ qh) {
    int grp = blockIdx.x * 4 + (threadIdx.x >> 5);   // 0..7
    v8f c[4] = {};
    wmma_tile_nt<4>(query_h, 2 * HH, 0, Wq_t, 2 * HH, grp * 64, 2 * HH, c);
    const int lane = threadIdx.x & 31, half = lane >> 4;
#pragma unroll
    for (int j = 0; j < 4; ++j) {
        int n = grp * 64 + 16 * j + (lane & 15);
#pragma unroll
        for (int r = 0; r < 8; ++r) {
            int m = r + 8 * half;
            qh[m * HH + n] = c[j][r] + bq[n];
        }
    }
}

// S3: per-batch attention: scores (softmax over L), ctx, mean-head scores out
__global__ void attn_kernel(const float* __restrict__ qh,
                            const float* __restrict__ Kp, const float* __restrict__ Vp,
                            float* __restrict__ ctx, h16* __restrict__ ctx_h,
                            float* __restrict__ sc_out,
                            const int* __restrict__ output_len, int t) {
    const int b = blockIdx.x, l = threadIdx.x;          // 128 threads = L
    __shared__ float s[LM];
    __shared__ float red[LM];
    const float scale = 0.125f;                         // 1/sqrt(64)
    float scAcc = 0.f;
    for (int h = 0; h < NHEAD; ++h) {
        const float* q  = qh + b * HH + h * DK;
        const float* kr = Kp + ((size_t)(l * BS + b)) * HH + h * DK;
        float d = 0.f;
#pragma unroll 8
        for (int j = 0; j < DK; ++j) d += q[j] * kr[j];
        d *= scale;
        red[l] = d; __syncthreads();
        for (int off = 64; off; off >>= 1) {
            if (l < off) red[l] = fmaxf(red[l], red[l + off]);
            __syncthreads();
        }
        float mx = red[0]; __syncthreads();
        float e = expf(d - mx);
        red[l] = e; __syncthreads();
        for (int off = 64; off; off >>= 1) {
            if (l < off) red[l] += red[l + off];
            __syncthreads();
        }
        float p = e / red[0];
        s[l] = p; __syncthreads();
        scAcc += p * (1.0f / NHEAD);
        if (l < DK) {
            float acc = 0.f;
            for (int ll = 0; ll < LM; ++ll)
                acc += s[ll] * Vp[((size_t)(ll * BS + b)) * HH + h * DK + l];
            ctx[b * HH + h * DK + l]   = acc;
            ctx_h[b * HH + h * DK + l] = f2h(acc);
        }
        __syncthreads();
    }
    const float active = (t < output_len[b]) ? 1.f : 0.f;
    sc_out[b * LM + l] = scAcc * active;
}

// S4: gi = ctx @ W_ih + b_ih ; gh = hid @ W_hh + b_hh  (M=16, K=512, N=1536 each)
// 2 mats x 24 N-groups (4 tiles each) = 48 waves.
__global__ void gru_mm_kernel(const h16* __restrict__ ctx_h,
                              const h16* __restrict__ query_h,   // cols 512..1023 hold f16(hid)
                              const h16* __restrict__ Wih_t, const h16* __restrict__ Whh_t,
                              const float* __restrict__ b_ih, const float* __restrict__ b_hh,
                              float* __restrict__ gi, float* __restrict__ gh) {
    int w   = blockIdx.x * 4 + (threadIdx.x >> 5);   // 0..47
    int mat = w / 24;
    int ng  = w % 24;
    const h16*   A    = mat ? (query_h + HH) : ctx_h;
    int          lda  = mat ? 2 * HH : HH;
    const h16*   Bt   = mat ? Whh_t : Wih_t;
    const float* bias = mat ? b_hh  : b_ih;
    float*       Out  = mat ? gh    : gi;
    v8f c[4] = {};
    wmma_tile_nt<4>(A, lda, 0, Bt, HH, ng * 64, HH, c);
    const int lane = threadIdx.x & 31, half = lane >> 4;
#pragma unroll
    for (int j = 0; j < 4; ++j) {
        int n = ng * 64 + 16 * j + (lane & 15);
#pragma unroll
        for (int r = 0; r < 8; ++r) {
            int m = r + 8 * half;
            Out[m * (3 * HH) + n] = c[j][r] + bias[n];
        }
    }
}

// S5: GRU gate elementwise; updates carry hid (unmasked), emits masked hiddens
__global__ void gru_gates_kernel(const float* __restrict__ gi, const float* __restrict__ gh,
                                 float* __restrict__ hid, h16* __restrict__ hnew_h,
                                 float* __restrict__ hid_out,
                                 const int* __restrict__ output_len, int t) {
    int i = blockIdx.x * 256 + threadIdx.x;          // 8192 threads
    if (i >= BS * HH) return;
    int b = i >> 9, j = i & (HH - 1);
    float ir = gi[b * 3 * HH + j],          hr = gh[b * 3 * HH + j];
    float iz = gi[b * 3 * HH + HH + j],     hz = gh[b * 3 * HH + HH + j];
    float in = gi[b * 3 * HH + 2 * HH + j], hn = gh[b * 3 * HH + 2 * HH + j];
    float r = 1.f / (1.f + expf(-(ir + hr)));
    float z = 1.f / (1.f + expf(-(iz + hz)));
    float n = tanhf(in + r * hn);
    float h = hid[i];
    float hnew = (1.f - z) * n + z * h;
    hid[i]    = hnew;          // carry, unmasked (matches lax.scan carry)
    hnew_h[i] = f2h(hnew);
    const float active = (t < output_len[b]) ? 1.f : 0.f;
    hid_out[i] = hnew * active;
}

// S6: logits = h_new @ Wf + bf  (M=16, K=512, N=16000)
// 250 waves x 4 tiles = 1000 tiles.
__global__ void logits_kernel(const h16* __restrict__ hnew_h,
                              const h16* __restrict__ Wf_t,
                              const float* __restrict__ bf_bias,
                              float* __restrict__ logits_cur,   // unmasked carry
                              float* __restrict__ out_logits,   // masked output slice
                              const int* __restrict__ output_len, int t) {
    int grp = blockIdx.x * 4 + (threadIdx.x >> 5);   // 0..251
    if (grp >= 250) return;
    int n0 = grp * 64;
    v8f c[4] = {};
    wmma_tile_nt<4>(hnew_h, HH, 0, Wf_t, HH, n0, HH, c);
    const int lane = threadIdx.x & 31, half = lane >> 4;
#pragma unroll
    for (int j = 0; j < 4; ++j) {
        int n = n0 + 16 * j + (lane & 15);
#pragma unroll
        for (int r = 0; r < 8; ++r) {
            int m = r + 8 * half;
            float val = c[j][r] + bf_bias[n];
            logits_cur[(size_t)m * VV + n] = val;
            const float active = (t < output_len[m]) ? 1.f : 0.f;
            out_logits[(size_t)m * VV + n] = val * active;
        }
    }
}

// ---------------------------------------------------------------------------
// Host launcher
// ---------------------------------------------------------------------------
extern "C" void kernel_launch(void* const* d_in, const int* in_sizes, int n_in,
                              void* d_out, int out_size, void* d_ws, size_t ws_size,
                              hipStream_t stream) {
    const float* memory = (const float*)d_in[0];
    const float* emb    = (const float*)d_in[1];
    const float* Wq     = (const float*)d_in[2];
    const float* bq     = (const float*)d_in[3];
    const float* Wk     = (const float*)d_in[4];
    const float* bk     = (const float*)d_in[5];
    const float* Wv     = (const float*)d_in[6];
    const float* bv     = (const float*)d_in[7];
    const float* W_ih   = (const float*)d_in[8];
    const float* b_ih   = (const float*)d_in[9];
    const float* W_hh   = (const float*)d_in[10];
    const float* b_hh   = (const float*)d_in[11];
    const float* Wf     = (const float*)d_in[12];
    const float* bf     = (const float*)d_in[13];
    const int*   olen   = (const int*)d_in[14];

    // workspace carve-up (256B aligned)
    char* ws = (char*)d_ws;
    size_t off = 0;
    auto alloc = [&](size_t bytes) -> char* {
        char* p = ws + off;
        off += (bytes + 255) & ~(size_t)255;
        return p;
    };
    h16*   Wq_t   = (h16*)  alloc((size_t)HH * 2 * HH * 2);
    h16*   Wih_t  = (h16*)  alloc((size_t)3 * HH * HH * 2);
    h16*   Whh_t  = (h16*)  alloc((size_t)3 * HH * HH * 2);
    h16*   Wf_t   = (h16*)  alloc((size_t)VV * HH * 2);
    h16*   Wk_t   = (h16*)  alloc((size_t)HH * DD * 2);
    h16*   Wv_t   = (h16*)  alloc((size_t)HH * DD * 2);
    h16*   mem_h  = (h16*)  alloc((size_t)LM * BS * DD * 2);
    float* Kp     = (float*)alloc((size_t)LM * BS * HH * 4);
    float* Vp     = (float*)alloc((size_t)LM * BS * HH * 4);
    float* logits = (float*)alloc((size_t)BS * VV * 4);
    float* hid    = (float*)alloc((size_t)BS * HH * 4);
    h16*   query  = (h16*)  alloc((size_t)BS * 2 * HH * 2);
    float* qh     = (float*)alloc((size_t)BS * HH * 4);
    float* ctx    = (float*)alloc((size_t)BS * HH * 4);
    h16*   ctx_h  = (h16*)  alloc((size_t)BS * HH * 2);
    float* gi     = (float*)alloc((size_t)BS * 3 * HH * 4);
    float* gh     = (float*)alloc((size_t)BS * 3 * HH * 4);
    h16*   hnew_h = (h16*)  alloc((size_t)BS * HH * 2);
    (void)ws_size; (void)n_in; (void)in_sizes; (void)out_size;

    float* out_f       = (float*)d_out;
    float* out_logits0 = out_f;                                   // (T,B,V)
    float* out_hid0    = out_f + (size_t)TT * BS * VV;            // (T,B,H)
    float* out_sc0     = out_f + (size_t)TT * BS * VV + (size_t)TT * BS * HH; // (T,B,L)

    // ---- prep: transpose/convert weights to f16 column-major ----
    auto transpose = [&](const float* W, h16* Wt, int K, int N) {
        size_t total = (size_t)K * N;
        int blocks = (int)((total + 255) / 256);
        transpose_f16_kernel<<<blocks, 256, 0, stream>>>(W, Wt, K, N);
    };
    transpose(Wq,   Wq_t,  2 * HH, HH);
    transpose(W_ih, Wih_t, HH,     3 * HH);
    transpose(W_hh, Whh_t, HH,     3 * HH);
    transpose(Wf,   Wf_t,  HH,     VV);
    transpose(Wk,   Wk_t,  DD,     HH);
    transpose(Wv,   Wv_t,  DD,     HH);

    {
        size_t n = (size_t)LM * BS * DD;
        convert_f16_kernel<<<(int)((n + 255) / 256), 256, 0, stream>>>(memory, mem_h, n);
    }
    zero_state_kernel<<<(BS * VV + 255) / 256, 256, 0, stream>>>(logits, hid);

    // ---- hoisted K/V projection (loop-invariant in the reference) ----
    kv_proj_kernel<<<512, 128, 0, stream>>>(mem_h, Wk_t, Wv_t, bk, bv, Kp, Vp);

    // ---- sequential decode loop ----
    for (int t = 0; t < TT; ++t) {
        argmax_embed_kernel<<<BS, 256, 0, stream>>>(logits, emb, hid, query);
        qproj_kernel<<<2, 128, 0, stream>>>(query, Wq_t, bq, qh);
        attn_kernel<<<BS, LM, 0, stream>>>(qh, Kp, Vp, ctx, ctx_h,
                                           out_sc0 + (size_t)t * BS * LM, olen, t);
        gru_mm_kernel<<<12, 128, 0, stream>>>(ctx_h, query, Wih_t, Whh_t,
                                              b_ih, b_hh, gi, gh);
        gru_gates_kernel<<<(BS * HH + 255) / 256, 256, 0, stream>>>(
            gi, gh, hid, hnew_h, out_hid0 + (size_t)t * BS * HH, olen, t);
        logits_kernel<<<63, 128, 0, stream>>>(hnew_h, Wf_t, bf, logits,
                                              out_logits0 + (size_t)t * BS * VV, olen, t);
    }
}
